// LSTMTagger_13683765805761
// MI455X (gfx1250) — matrix operand order repaired and checked by reading, not compiled, not run on hardware
//
#include <hip/hip_runtime.h>
#include <hip/hip_bf16.h>
#include <stddef.h>

// ---------------------------------------------------------------------------
// Problem constants (from reference): P=2048 paths, L=64, D=256, H=512, OUT=1
// ---------------------------------------------------------------------------
#define P_TOT 2048
#define L_SEQ 64
#define D_IN  256
#define H_DIM 512
#define G4H   2048          // 4*H
#define PB    16            // paths per workgroup (M dimension of WMMA tiles)
#define WAVES 32
#define BLK   (WAVES * 32)  // 1024 threads

typedef __attribute__((ext_vector_type(16))) __bf16          v16bf;
typedef __attribute__((ext_vector_type(8)))  float           v8f;
typedef __attribute__((ext_vector_type(4)))  unsigned short  u16x4;
typedef __attribute__((ext_vector_type(8)))  unsigned short  u16x8;
typedef __attribute__((ext_vector_type(16))) unsigned short  u16x16;

__device__ __forceinline__ unsigned short f2bf(float f) {
  unsigned u = __float_as_uint(f);
  u += 0x7FFFu + ((u >> 16) & 1u);      // round-to-nearest-even
  return (unsigned short)(u >> 16);
}
__device__ __forceinline__ unsigned fenc(float f) {   // order-preserving uint encoding
  unsigned u = __float_as_uint(f);
  return (u & 0x80000000u) ? ~u : (u | 0x80000000u);
}
__device__ __forceinline__ float fdec(unsigned e) {
  unsigned b = (e & 0x80000000u) ? (e & 0x7FFFFFFFu) : ~e;
  return __uint_as_float(b);
}
__device__ __forceinline__ float sigmoidf(float x) { return 1.0f / (1.0f + __expf(-x)); }

// A-matrix (16x32 bf16) per-lane layout: lanes 0-15 hold K[0..15] of
// row M=lane; lanes 16-31 hold K[16..31] of row M=lane-16.
// aOff is the per-lane element offset of chunk0 for the current K-step.
__device__ __forceinline__ v16bf load_tileA(const unsigned short* base, int aOff) {
  const u16x8 lo = *(const u16x8*)(base + aOff);
  const u16x8 hi = *(const u16x8*)(base + aOff + 8);
  u16x16 r = __builtin_shufflevector(lo, hi, 0,1,2,3,4,5,6,7,8,9,10,11,12,13,14,15);
  return __builtin_bit_cast(v16bf, r);
}
// B-matrix (32x16 bf16): lane n<16 holds K[0..15] of column n (= weight row n,
// K-contiguous); lane n+16 holds K[16..31]. One aligned 32B load per lane.
// Wk is a UNIFORM pointer (SGPR pair, advanced by the K loop); bFix is the
// loop-invariant per-lane element offset -> GVS addressing (saddr + voffset),
// zero per-iteration VALU address math.
__device__ __forceinline__ v16bf load_tileB(const unsigned short* Wk, int bFix) {
  const u16x16 r = *(const u16x16*)(Wk + bFix);
  return __builtin_bit_cast(v16bf, r);
}

// ---------------------------------------------------------------------------
// Kernel 0: fp32 weights -> bf16 workspace, fused bias, init pooled-max buffer
// ---------------------------------------------------------------------------
__global__ __launch_bounds__(256) void prep_kernel(
    const float* __restrict__ Wih, const float* __restrict__ Whh,
    const float* __restrict__ bih, const float* __restrict__ bhh,
    unsigned short* __restrict__ WihB, unsigned short* __restrict__ WhhB,
    float* __restrict__ bias, unsigned* __restrict__ pooledEnc) {
  const int i = blockIdx.x * blockDim.x + threadIdx.x;
  if (i < G4H * D_IN)  WihB[i] = f2bf(Wih[i]);
  if (i < G4H * H_DIM) WhhB[i] = f2bf(Whh[i]);
  if (i < G4H)         bias[i] = bih[i] + bhh[i];
  if (i < H_DIM)       pooledEnc[i] = 0u;   // encodes -NaN: identity for max
}

// ---------------------------------------------------------------------------
// Kernel 1: persistent fused LSTM. One workgroup = 16 paths, 32 waves.
// Wave w owns hidden units [16w, 16w+16): 4 C-tiles = {i,f,g,o} x 16 columns.
// ---------------------------------------------------------------------------
__global__ __launch_bounds__(BLK) void lstm_kernel(
    const int* __restrict__ paths, const int* __restrict__ lengths,
    const float* __restrict__ emb,
    const unsigned short* __restrict__ WihB, const unsigned short* __restrict__ WhhB,
    const float* __restrict__ bias, unsigned* __restrict__ pooledEnc) {
  __shared__ unsigned short xbuf[PB * D_IN];   // x_t  bf16, 8 KB
  __shared__ unsigned short hbuf[PB * H_DIM];  // h_t  bf16, 16 KB
  __shared__ int lenS[PB];

  const int tid  = threadIdx.x;
  const int lane = tid & 31;
  const int w    = tid >> 5;            // wave id 0..31
  const int pblk = blockIdx.x * PB;
  const int m    = lane & 15;           // A row / C column within tile
  const int half = lane >> 4;
  const int mhalf = half * 8;           // C/D row offset of this lane half

  if (tid < PB) lenS[tid] = lengths[pblk + tid];
  for (int i = tid; i < PB * H_DIM; i += BLK) hbuf[i] = 0;  // h0 = 0

  v8f acc[4], hreg, creg;
#pragma unroll
  for (int r = 0; r < 8; ++r) { hreg[r] = 0.f; creg[r] = 0.f; }

  float bval[4];
#pragma unroll
  for (int G = 0; G < 4; ++G)
    bval[G] = bias[G * H_DIM + w * 16 + m];

  // Loop-invariant per-lane element offsets (the K-base advances uniformly).
  const int aX0 = m * D_IN  + half * 16;                      // A from xbuf
  const int aH0 = m * H_DIM + half * 16;                      // A from hbuf
  int bIhF[4], bHhF[4];
#pragma unroll
  for (int G = 0; G < 4; ++G) {
    const int row = G * H_DIM + w * 16 + m;                   // weight row = N column
    bIhF[G] = row * D_IN  + half * 16;
    bHhF[G] = row * H_DIM + half * 16;
  }

  for (int t = 0; t < L_SEQ; ++t) {
    __syncthreads();   // h_{t-1} stores visible; x_{t-1} readers finished
    {                  // gather+convert x_t: thread -> path tid>>6, 4 dims each
      const int p     = tid >> 6;
      const int dbase = (tid & 63) * 4;
      const int node  = paths[(size_t)(pblk + p) * L_SEQ + t];
      const float4* s4 = (const float4*)(emb + (size_t)node * D_IN + dbase);
      float4 v0 = s4[0];
      u16x4 xs;
      xs[0] = f2bf(v0.x); xs[1] = f2bf(v0.y); xs[2] = f2bf(v0.z); xs[3] = f2bf(v0.w);
      *(u16x4*)(xbuf + p * D_IN + dbase) = xs;
    }
    __syncthreads();   // x_t ready

#pragma unroll
    for (int q = 0; q < 4; ++q)
#pragma unroll
      for (int r = 0; r < 8; ++r) acc[q][r] = 0.f;

    // gates += x_t @ W_ih^T   (K = 256)
    {
      const unsigned short* Wk = WihB;   // uniform; +32 elems per K-step (SALU)
      int aOff = aX0;
#pragma unroll 1
      for (int kb = 0; kb < D_IN; kb += 32) {
        v16bf a = load_tileA(xbuf, aOff);
        acc[0] = __builtin_amdgcn_wmma_f32_16x16x32_bf16(
            false, a, false, load_tileB(Wk, bIhF[0]), (short)0, acc[0], false, false);
        acc[1] = __builtin_amdgcn_wmma_f32_16x16x32_bf16(
            false, a, false, load_tileB(Wk, bIhF[1]), (short)0, acc[1], false, false);
        acc[2] = __builtin_amdgcn_wmma_f32_16x16x32_bf16(
            false, a, false, load_tileB(Wk, bIhF[2]), (short)0, acc[2], false, false);
        acc[3] = __builtin_amdgcn_wmma_f32_16x16x32_bf16(
            false, a, false, load_tileB(Wk, bIhF[3]), (short)0, acc[3], false, false);
        aOff += 32; Wk += 32;
      }
    }
    // gates += h_{t-1} @ W_hh^T   (K = 512)
    {
      const unsigned short* Wk = WhhB;   // uniform; +32 elems per K-step (SALU)
      int aOff = aH0;
#pragma unroll 1
      for (int kb = 0; kb < H_DIM; kb += 32) {
        v16bf a = load_tileA(hbuf, aOff);
        acc[0] = __builtin_amdgcn_wmma_f32_16x16x32_bf16(
            false, a, false, load_tileB(Wk, bHhF[0]), (short)0, acc[0], false, false);
        acc[1] = __builtin_amdgcn_wmma_f32_16x16x32_bf16(
            false, a, false, load_tileB(Wk, bHhF[1]), (short)0, acc[1], false, false);
        acc[2] = __builtin_amdgcn_wmma_f32_16x16x32_bf16(
            false, a, false, load_tileB(Wk, bHhF[2]), (short)0, acc[2], false, false);
        acc[3] = __builtin_amdgcn_wmma_f32_16x16x32_bf16(
            false, a, false, load_tileB(Wk, bHhF[3]), (short)0, acc[3], false, false);
        aOff += 32; Wk += 32;
      }
    }
    __syncthreads();   // all waves finished reading hbuf; safe to overwrite

    // Elementwise LSTM cell entirely in registers (fp32 state), masked update.
    // C/D tile layout: VGPR r = path row (mhalf+r), column = m + 16w.
#pragma unroll
    for (int r = 0; r < 8; ++r) {
      float iv = sigmoidf(acc[0][r] + bval[0]);
      float fv = sigmoidf(acc[1][r] + bval[1]);
      float gv = tanhf   (acc[2][r] + bval[2]);
      float ov = sigmoidf(acc[3][r] + bval[3]);
      float cn = fv * creg[r] + iv * gv;
      float hn = ov * tanhf(cn);
      const bool live = (t < lenS[mhalf + r]);
      creg[r] = live ? cn : creg[r];
      hreg[r] = live ? hn : hreg[r];
      hbuf[(mhalf + r) * H_DIM + w * 16 + m] = f2bf(hreg[r]);
    }
  }

  // Max-pool over this block's 16 paths, then global atomic max per hidden unit.
  {
    float mx = hreg[0];
#pragma unroll
    for (int r = 1; r < 8; ++r) mx = fmaxf(mx, hreg[r]);
    mx = fmaxf(mx, __shfl_xor(mx, 16, 32));   // combine lane halves (rows 0-7 / 8-15)
    if (lane < 16)
      atomicMax(&pooledEnc[w * 16 + lane], fenc(mx));
  }
}

// ---------------------------------------------------------------------------
// Kernel 2: out = sigmoid(pooled . W_lin + b_lin)
// ---------------------------------------------------------------------------
__global__ __launch_bounds__(H_DIM) void head_kernel(
    const unsigned* __restrict__ pooledEnc, const float* __restrict__ Wlin,
    const float* __restrict__ blin, float* __restrict__ out) {
  __shared__ float red[H_DIM];
  const int i = threadIdx.x;
  red[i] = fdec(pooledEnc[i]) * Wlin[i];
  __syncthreads();
  for (int s = H_DIM / 2; s > 0; s >>= 1) {
    if (i < s) red[i] += red[i + s];
    __syncthreads();
  }
  if (i == 0) out[0] = sigmoidf(red[0] + blin[0]);
}

// ---------------------------------------------------------------------------
// Launch: prep -> persistent LSTM (P/16 = 128 blocks) -> head
// ---------------------------------------------------------------------------
extern "C" void kernel_launch(void* const* d_in, const int* in_sizes, int n_in,
                              void* d_out, int out_size, void* d_ws, size_t ws_size,
                              hipStream_t stream) {
  const int*   paths   = (const int*)d_in[0];
  const int*   lengths = (const int*)d_in[1];
  const float* emb     = (const float*)d_in[2];
  const float* Wih     = (const float*)d_in[3];
  const float* Whh     = (const float*)d_in[4];
  const float* bih     = (const float*)d_in[5];
  const float* bhh     = (const float*)d_in[6];
  const float* Wlin    = (const float*)d_in[7];
  const float* blin    = (const float*)d_in[8];
  float*       out     = (float*)d_out;

  char* ws = (char*)d_ws;
  unsigned short* WihB   = (unsigned short*)(ws);                      // 1 MB
  unsigned short* WhhB   = (unsigned short*)(ws + (1u << 20));         // 2 MB
  float*          bias   = (float*)(ws + 3u * (1u << 20));             // 8 KB
  unsigned*       pooled = (unsigned*)(ws + 3u * (1u << 20) + 8192u);  // 2 KB

  prep_kernel<<<(G4H * H_DIM + 255) / 256, 256, 0, stream>>>(
      Wih, Whh, bih, bhh, WihB, WhhB, bias, pooled);
  lstm_kernel<<<P_TOT / PB, BLK, 0, stream>>>(
      paths, lengths, emb, WihB, WhhB, bias, pooled);
  head_kernel<<<1, H_DIM, 0, stream>>>(pooled, Wlin, blin, out);
}
